// EncoderLayer_66039417143498
// MI455X (gfx1250) — matrix operand order repaired and checked by reading, not compiled, not run on hardware
//
#include <hip/hip_runtime.h>
#include <hip/hip_bf16.h>

// MI455X / gfx1250: wave32, WMMA 16x16x32 bf16 (fp32 accum).
// Memory-bound layer (~435 GFLOP vs ~1 GB traffic @ 23.3 TB/s):
//  - bf16 operands halve activation bytes; fp32 accumulation via v_wmma_f32_16x16x32_bf16
//  - weights pre-transposed (N x K, bf16, 24 MB total -> resident in 192 MB L2)
//  - GEMM: A tile staged to LDS by the Tensor Data Mover (double-buffered,
//    s_wait_tensorcnt), B fragments stream straight from L2 in WMMA layout
//  - flash-attention with register-resident online softmax (C-layout row striping)

typedef __attribute__((ext_vector_type(16))) __bf16 v16bf;
typedef __attribute__((ext_vector_type(8)))  __bf16 v8bf;
typedef __attribute__((ext_vector_type(8)))  float  v8f;
typedef unsigned int u32x4 __attribute__((ext_vector_type(4)));
typedef int          i32x4 __attribute__((ext_vector_type(4)));
typedef int          i32x8 __attribute__((ext_vector_type(8)));

#define D_MODEL 1024
#define SEQ     4096
#define BATCH   4
#define NHEAD   16
#define HDIM    64
#define HLOC    8
#define STRIDE_ 128
#define NB      (SEQ / STRIDE_)      // 32 blocks per sequence
#define CCOL    32
#define DFF_    4096
#define TOKENS  (BATCH * SEQ)        // 16384
#define ROWP    40                   // padded LDS row (32 data + 8 pad bf16) -> 80B

#if defined(__gfx1250__) && __has_builtin(__builtin_amdgcn_tensor_load_to_lds) && \
    __has_builtin(__builtin_amdgcn_s_wait_tensorcnt)
#define USE_TDM 1
#else
#define USE_TDM 0
#endif

static __device__ __forceinline__ v8f zero8() {
  v8f z;
#pragma unroll
  for (int i = 0; i < 8; ++i) z[i] = 0.0f;
  return z;
}

// A-matrix fragment (16-bit, 16x32): lane m = lane%16, half = lane/16.
// elements 0..7  -> K = half*16 + j       (16B contiguous)
// elements 8..15 -> K = half*16 + 8 + j   (16B contiguous)
static __device__ __forceinline__ v16bf load_afrag(const __bf16* rowptr, int half) {
  v8bf lo = *(const v8bf*)(rowptr + half * 16);
  v8bf hi = *(const v8bf*)(rowptr + half * 16 + 8);
  v16bf a;
#pragma unroll
  for (int j = 0; j < 8; ++j) { a[j] = lo[j]; a[8 + j] = hi[j]; }
  return a;
}

// ---------------------------------------------------------------------------
// Generic bf16 GEMM:  C[M,N] = A[M,K] * Bt[N,K]^T + bias.
// Block 256 threads = 8 waves (4 M x 2 N); block tile 128x128, wave tile
// 32x64 = 2x4 WMMA tiles.  A tile (128x32) is staged to LDS (TDM when
// available, padded rows for conflict-free ds_load_b128), double-buffered.
// Bt rows are contiguous in K -> B fragments load straight from L2.
// ---------------------------------------------------------------------------
template <bool F32OUT, bool BF16OUT, bool RELU>
__global__ __launch_bounds__(256)
void gemm_bf16_kernel(const __bf16* __restrict__ A, const __bf16* __restrict__ Bt,
                      const float* __restrict__ bias, float* __restrict__ Cf,
                      __bf16* __restrict__ Cb, int M, int N, int K)
{
  __shared__ __bf16 As[2][128 * ROWP];   // 2 x 10 KB

  const int tid  = threadIdx.x;
  const int lane = tid & 31, wave = tid >> 5;
  const int half = lane >> 4, ln = lane & 15;
  const int waveM = wave & 3, waveN = wave >> 2;
  const int mBlock = blockIdx.y * 128;
  const int mBase  = mBlock + waveM * 32;
  const int nBase  = blockIdx.x * 128 + waveN * 64;

  // ---- A tile staging (TDM or cooperative fallback) ----
  auto stage = [&](int buf, int k0) {
#if USE_TDM
    if (wave == 0) {
      unsigned long long ga =
          (unsigned long long)(A + (size_t)mBlock * K + k0);   // byte address
      unsigned lds = (unsigned)(unsigned long long)(void*)&As[buf][0];
      u32x4 g0;
      g0[0] = 1u;                                  // count=1, user descriptor
      g0[1] = lds;                                 // LDS byte address
      g0[2] = (unsigned)ga;                        // global_addr[31:0]
      g0[3] = (unsigned)((ga >> 32) & 0x1FFFFFFull) | (2u << 30);  // [56:32] | type=2
      i32x8 g1;
      g1[0] = (1 << 16)      // data_size = 2 bytes
            | (1 << 20)      // pad_enable
            | (3 << 22)      // pad_interval: 16 DWORDs (64B row)
            | (3 << 25);     // pad_amount:   4 DWORDs (16B = 8 bf16)
      g1[1] = (K & 0xFFFF) << 16;                            // tensor_dim0 lo
      g1[2] = ((K >> 16) & 0xFFFF) | ((M & 0xFFFF) << 16);   // dim0 hi | dim1 lo
      g1[3] = ((M >> 16) & 0xFFFF) | (32 << 16);             // dim1 hi | tile_dim0=32
      g1[4] = 128;                                           // tile_dim1=128, tile_dim2=0
      g1[5] = K;                                             // dim0_stride lo32 (elems)
      g1[6] = 0;
      g1[7] = 0;
      i32x4 gz;
      gz[0] = 0; gz[1] = 0; gz[2] = 0; gz[3] = 0;
      i32x8 gz8;
#pragma unroll
      for (int z = 0; z < 8; ++z) gz8[z] = 0;
      __builtin_amdgcn_tensor_load_to_lds(g0, g1, gz, gz, gz8, 0);
    }
#else
    int row = tid >> 1;
    int seg = (tid & 1) * 16;
    const __bf16* src = A + (size_t)(mBlock + row) * K + k0 + seg;
    v8bf x0 = *(const v8bf*)(src);
    v8bf x1 = *(const v8bf*)(src + 8);
    *(v8bf*)(&As[buf][row * ROWP + seg])     = x0;
    *(v8bf*)(&As[buf][row * ROWP + seg + 8]) = x1;
#endif
  };

  v8f acc[2][4];
#pragma unroll
  for (int i = 0; i < 2; ++i)
#pragma unroll
    for (int j = 0; j < 4; ++j) acc[i][j] = zero8();

  int cur = 0;
  stage(0, 0);
  for (int k0 = 0; k0 < K; k0 += 32) {
    if (k0 + 32 < K) {
      stage(cur ^ 1, k0 + 32);
#if USE_TDM
      if (wave == 0) __builtin_amdgcn_s_wait_tensorcnt(1);
#endif
    } else {
#if USE_TDM
      if (wave == 0) __builtin_amdgcn_s_wait_tensorcnt(0);
#endif
    }
    __syncthreads();

    v16bf a[2];
#pragma unroll
    for (int ms = 0; ms < 2; ++ms)
      a[ms] = load_afrag(&As[cur][(waveM * 32 + ms * 16 + ln) * ROWP], half);
    v16bf b[4];
#pragma unroll
    for (int ns = 0; ns < 4; ++ns)   // B frag: elements j -> K = half*16+j, 32B contiguous
      b[ns] = *(const v16bf*)(Bt + (size_t)(nBase + ns * 16 + ln) * K + k0 + half * 16);
#pragma unroll
    for (int ms = 0; ms < 2; ++ms)
#pragma unroll
      for (int ns = 0; ns < 4; ++ns)
        acc[ms][ns] = __builtin_amdgcn_wmma_f32_16x16x32_bf16(
            false, a[ms], false, b[ns], (short)0, acc[ms][ns], false, false);

    __syncthreads();
    cur ^= 1;
  }

  // epilogue: C layout row = r + 8*half, col = lane%16; 32-bit offsets
#pragma unroll
  for (int ms = 0; ms < 2; ++ms) {
    int rowb = mBase + ms * 16 + half * 8;
#pragma unroll
    for (int ns = 0; ns < 4; ++ns) {
      int col = nBase + ns * 16 + ln;
      float bcol = bias[col];
#pragma unroll
      for (int r = 0; r < 8; ++r) {
        float vv = acc[ms][ns][r] + bcol;
        if (RELU) vv = fmaxf(vv, 0.0f);
        int o = (rowb + r) * N + col;
        if (F32OUT)  Cf[o] = vv;
        if (BF16OUT) Cb[o] = (__bf16)vv;
      }
    }
  }
}

// ---------------------------------------------------------------------------
// Flash attention, templated:
//   GLOBAL=0: local block attention, heads 0..7, keys = own 128-token block
//   GLOBAL=1: global attention,      heads 8..15, keys = 1024 gathered
//             positions (pos % 128 >= 96), streamed in 8 tiles of 128
// Grid (32, 8, 4) = (q-tile, head, batch); 256 threads = 8 waves,
// each wave owns 16 query rows.  Online softmax kept in registers by
// exploiting the C-layout (full score row lives in one VGPR index across
// 16 lanes of a half -> 4 shfl_xor hops reduce it).
// ---------------------------------------------------------------------------
template <int GLOBAL>
__global__ __launch_bounds__(256)
void attn_kernel(const __bf16* __restrict__ q, const __bf16* __restrict__ k,
                 const __bf16* __restrict__ v, const unsigned char* __restrict__ mask,
                 __bf16* __restrict__ out)
{
  __shared__ __bf16 Pbuf[8][16][128];   // per-wave P tile (bf16)       32 KB
  __shared__ __bf16 VT[64][128];        // V tile transposed [hd][key]  16 KB

  const int qt = blockIdx.x;            // 0..31 (128-query tile)
  const int h  = blockIdx.y;            // 0..7
  const int b  = blockIdx.z;            // 0..3
  const int hoff = (GLOBAL ? (HLOC + h) : h) * HDIM;

  const int tid  = threadIdx.x;
  const int lane = tid & 31, wave = tid >> 5;
  const int half = lane >> 4, ln = lane & 15;

  // Q fragments for this wave's 16 rows (2 K-steps over HDIM=64)
  const __bf16* qrow = q + (size_t)(b * SEQ + qt * 128 + wave * 16 + ln) * D_MODEL + hoff;
  v16bf aq[2];
#pragma unroll
  for (int ks = 0; ks < 2; ++ks) aq[ks] = load_afrag(qrow + ks * 32, half);

  v8f o[4];
#pragma unroll
  for (int i = 0; i < 4; ++i) o[i] = zero8();
  float mrow[8], lrow[8];
#pragma unroll
  for (int r = 0; r < 8; ++r) { mrow[r] = -1e30f; lrow[r] = 0.0f; }

  const int nkt = GLOBAL ? 8 : 1;
  for (int kt = 0; kt < nkt; ++kt) {
    // ---- S = Q * K^T (scaled), 8 n-tiles x 2 k-steps ----
    v8f s[8];
#pragma unroll
    for (int i = 0; i < 8; ++i) s[i] = zero8();
    int posv[8];
#pragma unroll
    for (int nt = 0; nt < 8; ++nt) {
      int key = nt * 16 + ln;
      int pos;
      if (GLOBAL) { int g = kt * 128 + key; pos = (g >> 5) * STRIDE_ + (STRIDE_ - CCOL) + (g & 31); }
      else        { pos = qt * 128 + key; }
      posv[nt] = pos;
      const __bf16* krow = k + (size_t)(b * SEQ + pos) * D_MODEL + hoff;
#pragma unroll
      for (int ks = 0; ks < 2; ++ks) {
        v16bf bk = *(const v16bf*)(krow + ks * 32 + half * 16);  // B frag: contiguous K-row
        s[nt] = __builtin_amdgcn_wmma_f32_16x16x32_bf16(
            false, aq[ks], false, bk, (short)0, s[nt], false, false);
      }
    }
    // scale + mask (branchless)
#pragma unroll
    for (int nt = 0; nt < 8; ++nt) {
      bool keep = mask[(size_t)b * SEQ + posv[nt]] != 0;
#pragma unroll
      for (int r = 0; r < 8; ++r) {
        float sv = s[nt][r] * 0.125f;
        s[nt][r] = keep ? sv : -1e9f;
      }
    }
    // online softmax update (per row r of this wave's tile)
#pragma unroll
    for (int r = 0; r < 8; ++r) {
      float tmax = -1e30f;
#pragma unroll
      for (int nt = 0; nt < 8; ++nt) tmax = fmaxf(tmax, s[nt][r]);
#pragma unroll
      for (int off = 1; off < 16; off <<= 1) tmax = fmaxf(tmax, __shfl_xor(tmax, off, 32));
      float mnew = fmaxf(mrow[r], tmax);
      float corr = __expf(mrow[r] - mnew);
      mrow[r] = mnew;
      float psum = 0.0f;
#pragma unroll
      for (int nt = 0; nt < 8; ++nt) { float p = __expf(s[nt][r] - mnew); s[nt][r] = p; psum += p; }
#pragma unroll
      for (int off = 1; off < 16; off <<= 1) psum += __shfl_xor(psum, off, 32);
      lrow[r] = lrow[r] * corr + psum;
#pragma unroll
      for (int i = 0; i < 4; ++i) o[i][r] *= corr;
    }
    // P -> LDS (per-wave region, C layout -> row-major bf16)
#pragma unroll
    for (int nt = 0; nt < 8; ++nt)
#pragma unroll
      for (int r = 0; r < 8; ++r)
        Pbuf[wave][half * 8 + r][nt * 16 + ln] = (__bf16)s[nt][r];
    // stage V tile transposed: VT[d][key]
    {
      int j = tid >> 1;
      int dbase = (tid & 1) * 32;
      int pos;
      if (GLOBAL) { int g = kt * 128 + j; pos = (g >> 5) * STRIDE_ + (STRIDE_ - CCOL) + (g & 31); }
      else        { pos = qt * 128 + j; }
      const __bf16* vrow = v + (size_t)(b * SEQ + pos) * D_MODEL + hoff + dbase;
#pragma unroll
      for (int t = 0; t < 32; ++t) VT[dbase + t][j] = vrow[t];
    }
    __syncthreads();
    // ---- O += P * V  (K = 128 keys -> 4 k-steps; 4 n-tiles over HDIM) ----
#pragma unroll
    for (int ks2 = 0; ks2 < 4; ++ks2) {
      v16bf ap = load_afrag(&Pbuf[wave][ln][ks2 * 32], half);
#pragma unroll
      for (int nt2 = 0; nt2 < 4; ++nt2) {
        v16bf bv = *(const v16bf*)(&VT[nt2 * 16 + ln][ks2 * 32 + half * 16]);
        o[nt2] = __builtin_amdgcn_wmma_f32_16x16x32_bf16(
            false, ap, false, bv, (short)0, o[nt2], false, false);
      }
    }
    __syncthreads();
  }
  // normalize + store bf16 into concatenated attention output
#pragma unroll
  for (int r = 0; r < 8; ++r) {
    float invl = 1.0f / lrow[r];
    int row = qt * 128 + wave * 16 + half * 8 + r;
#pragma unroll
    for (int nt2 = 0; nt2 < 4; ++nt2) {
      int col = hoff + nt2 * 16 + ln;
      out[(size_t)(b * SEQ + row) * D_MODEL + col] = (__bf16)(o[nt2][r] * invl);
    }
  }
}

// ---------------------------------------------------------------------------
// residual add + LayerNorm over D=1024; one block per token row.
// ---------------------------------------------------------------------------
__global__ __launch_bounds__(256)
void add_ln_kernel(const float* __restrict__ x1, const float* __restrict__ x2,
                   const float* __restrict__ g, const float* __restrict__ beta,
                   float* __restrict__ outf, __bf16* __restrict__ outb)
{
  const int row = blockIdx.x;
  const int tid = threadIdx.x;
  const size_t base = (size_t)row * D_MODEL;
  float vals[4];
  float s = 0.f, ss = 0.f;
#pragma unroll
  for (int i = 0; i < 4; ++i) {
    int idx = tid * 4 + i;
    float t = x1[base + idx] + x2[base + idx];
    vals[i] = t; s += t; ss += t * t;
  }
#pragma unroll
  for (int off = 1; off < 32; off <<= 1) {
    s  += __shfl_xor(s,  off, 32);
    ss += __shfl_xor(ss, off, 32);
  }
  __shared__ float rs[8], rss[8];
  int wave = tid >> 5, lane = tid & 31;
  if (lane == 0) { rs[wave] = s; rss[wave] = ss; }
  __syncthreads();
  float ts = 0.f, tss = 0.f;
#pragma unroll
  for (int w = 0; w < 8; ++w) { ts += rs[w]; tss += rss[w]; }
  float mean = ts * (1.0f / D_MODEL);
  float var  = tss * (1.0f / D_MODEL) - mean * mean;
  float rstd = rsqrtf(var + 1e-6f);
#pragma unroll
  for (int i = 0; i < 4; ++i) {
    int idx = tid * 4 + i;
    float y = g[idx] * ((vals[i] - mean) * rstd) + beta[idx];
    outf[base + idx] = y;
    if (outb) outb[base + idx] = (__bf16)y;
  }
}

// fp32 -> bf16 convert (n multiple of 256)
__global__ __launch_bounds__(256)
void cvt_bf16_kernel(const float* __restrict__ in, __bf16* __restrict__ out, int n)
{
  int i = blockIdx.x * 256 + threadIdx.x;
  if (i < n) out[i] = (__bf16)in[i];
}

// fp32 [K,N] -> bf16 transposed [N,K]
__global__ __launch_bounds__(256)
void tcvt_bf16_kernel(const float* __restrict__ in, __bf16* __restrict__ out, int K, int N)
{
  int i = blockIdx.x * 256 + threadIdx.x;
  int kk = i / N, nn = i % N;
  out[(size_t)nn * K + kk] = (__bf16)in[i];
}

// ---------------------------------------------------------------------------
extern "C" void kernel_launch(void* const* d_in, const int* in_sizes, int n_in,
                              void* d_out, int out_size, void* d_ws, size_t ws_size,
                              hipStream_t stream)
{
  const float* src  = (const float*)d_in[0];
  const unsigned char* mask = (const unsigned char*)d_in[1];
  const float* Wq = (const float*)d_in[2];   const float* bq = (const float*)d_in[3];
  const float* Wk = (const float*)d_in[4];   const float* bk = (const float*)d_in[5];
  const float* Wv = (const float*)d_in[6];   const float* bv = (const float*)d_in[7];
  const float* Wo = (const float*)d_in[8];   const float* bo = (const float*)d_in[9];
  const float* ln1g = (const float*)d_in[10]; const float* ln1b = (const float*)d_in[11];
  const float* W1 = (const float*)d_in[12];  const float* b1 = (const float*)d_in[13];
  const float* W2 = (const float*)d_in[14];  const float* b2 = (const float*)d_in[15];
  const float* ln2g = (const float*)d_in[16]; const float* ln2b = (const float*)d_in[17];
  float* outp = (float*)d_out;

  // workspace carve-out
  char* cur = (char*)d_ws;
  auto alloc = [&](size_t bytes) -> void* {
    void* p = (void*)cur;
    cur += (bytes + 255) & ~(size_t)255;
    return p;
  };
  __bf16* src_bf  = (__bf16*)alloc((size_t)TOKENS * D_MODEL * 2);
  __bf16* WqT     = (__bf16*)alloc((size_t)D_MODEL * D_MODEL * 2);
  __bf16* WkT     = (__bf16*)alloc((size_t)D_MODEL * D_MODEL * 2);
  __bf16* WvT     = (__bf16*)alloc((size_t)D_MODEL * D_MODEL * 2);
  __bf16* WoT     = (__bf16*)alloc((size_t)D_MODEL * D_MODEL * 2);
  __bf16* W1T     = (__bf16*)alloc((size_t)D_MODEL * DFF_ * 2);
  __bf16* W2T     = (__bf16*)alloc((size_t)DFF_ * D_MODEL * 2);
  __bf16* q_bf    = (__bf16*)alloc((size_t)TOKENS * D_MODEL * 2);
  __bf16* k_bf    = (__bf16*)alloc((size_t)TOKENS * D_MODEL * 2);
  __bf16* v_bf    = (__bf16*)alloc((size_t)TOKENS * D_MODEL * 2);
  __bf16* attn_cat= (__bf16*)alloc((size_t)TOKENS * D_MODEL * 2);
  float*  oproj   = (float*) alloc((size_t)TOKENS * D_MODEL * 4);
  float*  attn_f  = (float*) alloc((size_t)TOKENS * D_MODEL * 4);
  __bf16* attn_b  = (__bf16*)alloc((size_t)TOKENS * D_MODEL * 2);
  __bf16* ffn1_bf = (__bf16*)alloc((size_t)TOKENS * DFF_ * 2);
  float*  ffn2_f  = (float*) alloc((size_t)TOKENS * D_MODEL * 4);

  dim3 blk(256);

  // 1) convert src + weights to bf16 (weights transposed to NxK)
  {
    int n = TOKENS * D_MODEL;
    cvt_bf16_kernel<<<n / 256, blk, 0, stream>>>(src, src_bf, n);
  }
  tcvt_bf16_kernel<<<(D_MODEL * D_MODEL) / 256, blk, 0, stream>>>(Wq, WqT, D_MODEL, D_MODEL);
  tcvt_bf16_kernel<<<(D_MODEL * D_MODEL) / 256, blk, 0, stream>>>(Wk, WkT, D_MODEL, D_MODEL);
  tcvt_bf16_kernel<<<(D_MODEL * D_MODEL) / 256, blk, 0, stream>>>(Wv, WvT, D_MODEL, D_MODEL);
  tcvt_bf16_kernel<<<(D_MODEL * D_MODEL) / 256, blk, 0, stream>>>(Wo, WoT, D_MODEL, D_MODEL);
  tcvt_bf16_kernel<<<(D_MODEL * DFF_) / 256, blk, 0, stream>>>(W1, W1T, D_MODEL, DFF_);
  tcvt_bf16_kernel<<<(DFF_ * D_MODEL) / 256, blk, 0, stream>>>(W2, W2T, DFF_, D_MODEL);

  // 2) QKV projections (bf16 out)
  dim3 gD(D_MODEL / 128, TOKENS / 128);      // (8, 128)
  gemm_bf16_kernel<false, true, false><<<gD, blk, 0, stream>>>(src_bf, WqT, bq, nullptr, q_bf, TOKENS, D_MODEL, D_MODEL);
  gemm_bf16_kernel<false, true, false><<<gD, blk, 0, stream>>>(src_bf, WkT, bk, nullptr, k_bf, TOKENS, D_MODEL, D_MODEL);
  gemm_bf16_kernel<false, true, false><<<gD, blk, 0, stream>>>(src_bf, WvT, bv, nullptr, v_bf, TOKENS, D_MODEL, D_MODEL);

  // 3) attention (local heads 0..7, global heads 8..15)
  dim3 ga(NB, HLOC, BATCH);                  // (32, 8, 4)
  attn_kernel<0><<<ga, blk, 0, stream>>>(q_bf, k_bf, v_bf, mask, attn_cat);
  attn_kernel<1><<<ga, blk, 0, stream>>>(q_bf, k_bf, v_bf, mask, attn_cat);

  // 4) output projection + residual + LN1
  gemm_bf16_kernel<true, false, false><<<gD, blk, 0, stream>>>(attn_cat, WoT, bo, oproj, nullptr, TOKENS, D_MODEL, D_MODEL);
  add_ln_kernel<<<TOKENS, blk, 0, stream>>>(src, oproj, ln1g, ln1b, attn_f, attn_b);

  // 5) FFN + residual + LN2
  dim3 gF(DFF_ / 128, TOKENS / 128);         // (32, 128)
  gemm_bf16_kernel<false, true, true><<<gF, blk, 0, stream>>>(attn_b, W1T, b1, nullptr, ffn1_bf, TOKENS, DFF_, D_MODEL);
  gemm_bf16_kernel<true, false, false><<<gD, blk, 0, stream>>>(ffn1_bf, W2T, b2, ffn2_f, nullptr, TOKENS, D_MODEL, DFF_);
  add_ln_kernel<<<TOKENS, blk, 0, stream>>>(attn_f, ffn2_f, ln2g, ln2b, outp, nullptr);

  (void)in_sizes; (void)n_in; (void)out_size; (void)ws_size;
}
